// CrossAttention_42769284333646
// MI455X (gfx1250) — matrix operand order repaired and checked by reading, not compiled, not run on hardware
//
#include <hip/hip_runtime.h>
#include <hip/hip_bf16.h>
#include <math.h>

typedef __attribute__((ext_vector_type(16))) _Float16 v16h;
typedef __attribute__((ext_vector_type(8)))  _Float16 v8h;
typedef __attribute__((ext_vector_type(8)))  float    v8f;

#define HH   64
#define WW   64
#define HW   4096
#define CC   256
#define NHD  8
#define NPT  4
#define HD   32
#define KD   256   // GEMM K dimension (== CC)

// ---------------------------------------------------------------------------
// Kernel 1: positional encoding + residual save + f16 transposed copies
//   q_f32[b,c,hw] = query + PE           (kept for the 2*q epilogue)
//   qf16[b,hw,c]  = (f16)(query + PE)    (A-matrix for offset/attn GEMMs)
//   vf16[b,hw,c]  = (f16) value          (A-matrix for value GEMM)
// ---------------------------------------------------------------------------
__global__ __launch_bounds__(256)
void prep_kernel(const float* __restrict__ query,
                 const float* __restrict__ value,
                 float* __restrict__ qf32,
                 _Float16* __restrict__ qf16,
                 _Float16* __restrict__ vf16,
                 int total) {
    int idx = blockIdx.x * blockDim.x + threadIdx.x;
    if (idx >= total) return;
    int pos = idx & (HW - 1);
    int c   = (idx >> 12) & (CC - 1);
    int b   = idx >> 20;
    int row = pos >> 6;          // y within H
    int col = pos & 63;          // x within W
    int k = c >> 2, j = c & 3;
    // div[k] = exp(2k * (-ln(1e4)/128))
    float divk = expf((float)(2 * k) * (-0.0719557841f));
    float p    = (j < 2) ? (float)(col + 1) : (float)(row + 1);
    float arg  = p * divk;
    float pe   = (j & 1) ? cosf(arg) : sinf(arg);
    float q = query[idx] + pe;
    qf32[idx] = q;
    int t = (b * HW + pos) * CC + c;
    qf16[t] = (_Float16)q;
    vf16[t] = (_Float16)value[idx];
}

// ---------------------------------------------------------------------------
// Kernel 2: pre-swizzle f32 weights [K=256, N] into WMMA B-fragment layout.
// Per (ntile, kstep): 32 lanes x 16 halfs, contiguous per lane, so the GEMM
// loads one 32B v16h per lane. B fragment (contiguous k order used here): lane = N-col (lane&15),
// lanes 16-31 hold K=16..31 of the step; VGPR j packs K=2j,2j+1.
// ---------------------------------------------------------------------------
__global__ __launch_bounds__(256)
void wconv_kernel(const float* __restrict__ w, _Float16* __restrict__ frag, int N) {
    int tid   = blockIdx.x * blockDim.x + threadIdx.x;
    int lane  = tid & 31;
    int t2    = tid >> 5;
    int kstep = t2 & 7;          // 8 steps of K=32
    int ntile = t2 >> 3;
    if (ntile * 16 >= N) return;
    int colh  = lane & 15;
    int khalf = lane >> 4;
    int col   = ntile * 16 + colh;
    _Float16* dst = frag + ((size_t)(ntile * 8 + kstep) * 32 + lane) * 16;
    int kg0 = kstep * 32 + khalf * 16;
#pragma unroll
    for (int i = 0; i < 16; ++i)
        dst[i] = (_Float16)w[(size_t)(kg0 + i) * N + col];
}

// ---------------------------------------------------------------------------
// WMMA GEMM core, register-blocked over N:
// Block = 4 waves; each wave owns one M-tile and NT accumulators covering
// NT consecutive N-tiles. The NT B-tile fragments (NT*8KB) are staged in LDS
// once per block; each K-step loads ONE A-fragment and issues NT WMMAs.
// A-fragment (16-bit A 16x32): lane row = lane&15; lanes 0-15 hold K 0..15
// of the step, lanes 16-31 hold K 16..31 -> two 16B loads.
// ---------------------------------------------------------------------------
template<int NT>
__device__ __forceinline__ void gemm_core(const _Float16* __restrict__ A,
                                          const _Float16* __restrict__ Bfrag,
                                          const float* __restrict__ bias,
                                          _Float16* lb, int ng0, int mtile,
                                          v8f (&acc)[NT]) {
    int lane = threadIdx.x & 31;
    // cooperative stage of NT B-tiles into LDS
    {
        const uint4* src = (const uint4*)(Bfrag + (size_t)ng0 * 8 * 32 * 16);
        uint4* dst = (uint4*)lb;
        for (int i = threadIdx.x; i < NT * 512; i += 128) dst[i] = src[i];
    }
    __syncthreads();
    int colh = lane & 15, khalf = lane >> 4;
    const _Float16* arow = A + (size_t)(mtile * 16 + colh) * KD;
#pragma unroll
    for (int t = 0; t < NT; ++t) {
        float bv = bias[(ng0 + t) * 16 + colh];
#pragma unroll
        for (int r = 0; r < 8; ++r) acc[t][r] = bv;
    }
#pragma unroll
    for (int ks = 0; ks < 8; ++ks) {
        int k0 = ks * 32;
        v8h alo = *(const v8h*)(arow + k0 + khalf * 16);
        v8h ahi = *(const v8h*)(arow + k0 + khalf * 16 + 8);
        v16h a  = __builtin_shufflevector(alo, ahi,
                     0,1,2,3,4,5,6,7,8,9,10,11,12,13,14,15);
#pragma unroll
        for (int t = 0; t < NT; ++t) {
            v16h bm = *(const v16h*)(lb + (size_t)((t * 8 + ks) * 32 + lane) * 16);
            acc[t] = __builtin_amdgcn_wmma_f32_16x16x32_f16(
                         false, a, false, bm, (short)0, acc[t], false, false);
        }
    }
}

template<int NT>
__global__ __launch_bounds__(128)
void gemm_kernel(const _Float16* __restrict__ A,
                 const _Float16* __restrict__ Bfrag,
                 const float* __restrict__ bias,
                 float* __restrict__ Out, int N) {
    __shared__ _Float16 lb[NT * 8 * 32 * 16];
    int wave = threadIdx.x >> 5, lane = threadIdx.x & 31;
    int ng0   = blockIdx.y * NT;
    int mtile = blockIdx.x * 4 + wave;
    v8f acc[NT];
    gemm_core<NT>(A, Bfrag, bias, lb, ng0, mtile, acc);
    int colh = lane & 15, khalf = lane >> 4;
    int m0 = mtile * 16 + khalf * 8;     // lane's 8 consecutive output rows
#pragma unroll
    for (int t = 0; t < NT; ++t) {
        int n = (ng0 + t) * 16 + colh;
#pragma unroll
        for (int r = 0; r < 8; ++r)
            Out[(size_t)(m0 + r) * N + n] = acc[t][r];
    }
}

// Final GEMM: out[b,c,hw] = sampled@w_out + b_out + 2*q   (qf residual + outer +q)
template<int NT>
__global__ __launch_bounds__(128)
void gemm_final_kernel(const _Float16* __restrict__ A,
                       const _Float16* __restrict__ Bfrag,
                       const float* __restrict__ bias,
                       const float* __restrict__ qf32,
                       float* __restrict__ Out) {
    __shared__ _Float16 lb[NT * 8 * 32 * 16];
    int wave = threadIdx.x >> 5, lane = threadIdx.x & 31;
    int ng0   = blockIdx.y * NT;
    int mtile = blockIdx.x * 4 + wave;
    v8f acc[NT];
    gemm_core<NT>(A, Bfrag, bias, lb, ng0, mtile, acc);
    int colh = lane & 15, khalf = lane >> 4;
    int m0  = mtile * 16 + khalf * 8;          // global row = b*HW + pos
    int b   = m0 >> 12;
    int pos = m0 & (HW - 1);
#pragma unroll
    for (int t = 0; t < NT; ++t) {
        int n = (ng0 + t) * 16 + colh;               // output channel c
        size_t obase = ((size_t)b * CC + n) * HW + pos;   // contiguous over r
#pragma unroll
        for (int r = 0; r < 8; ++r)
            Out[obase + r] = acc[t][r] + 2.0f * qf32[obase + r];
    }
}

// ---------------------------------------------------------------------------
// Kernel 4: softmax over 4 points + bilinear gather-weighted sum.
// One wave per (b, q, head); lane == head-dim element (HD==32==wave32).
// Each corner read is a coalesced 128B row of the projected value.
// ---------------------------------------------------------------------------
__global__ __launch_bounds__(256)
void sample_kernel(const float* __restrict__ vbuf,    // [B,HW,(h,d)] row-major
                   const float* __restrict__ offb,    // [B*HW, 64]
                   const float* __restrict__ attnb,   // [B*HW, 32]
                   _Float16* __restrict__ sampled) {  // [B*HW, C] f16
    int gid  = blockIdx.x * 8 + (threadIdx.x >> 5);
    int lane = threadIdx.x & 31;
    int h  = gid & 7;
    int qi = (gid >> 3) & (HW - 1);
    int b  = gid >> 15;
    const float* orow = offb  + (size_t)(b * HW + qi) * (NHD * NPT * 2) + h * NPT * 2;
    const float* lrow = attnb + (size_t)(b * HW + qi) * (NHD * NPT)     + h * NPT;
    float l0 = lrow[0], l1 = lrow[1], l2 = lrow[2], l3 = lrow[3];
    float m  = fmaxf(fmaxf(l0, l1), fmaxf(l2, l3));
    float e0 = __expf(l0 - m), e1 = __expf(l1 - m);
    float e2 = __expf(l2 - m), e3 = __expf(l3 - m);
    float inv = 1.0f / (e0 + e1 + e2 + e3);
    float aw[4] = { e0 * inv, e1 * inv, e2 * inv, e3 * inv };
    // reference_points quirk: ref = ((q % H)/(H-1), (q / H)/(W-1))
    float refx = (float)(qi & 63) * (1.0f / 63.0f);
    float refy = (float)(qi >> 6) * (1.0f / 63.0f);
    const float* vb = vbuf + (size_t)b * HW * CC + h * HD + lane;
    float acc = 0.0f;
#pragma unroll
    for (int p = 0; p < NPT; ++p) {
        float x = (refx + orow[2 * p    ] * (1.0f / WW)) * WW - 0.5f;
        float y = (refy + orow[2 * p + 1] * (1.0f / HH)) * HH - 0.5f;
        float x0f = floorf(x), y0f = floorf(y);
        float fx = x - x0f, fy = y - y0f;
        int x0 = (int)x0f, y0 = (int)y0f;
        float sp = 0.0f;
#pragma unroll
        for (int dy = 0; dy < 2; ++dy)
#pragma unroll
            for (int dx = 0; dx < 2; ++dx) {
                int xi = x0 + dx, yi = y0 + dy;
                if (xi >= 0 && xi < WW && yi >= 0 && yi < HH) {
                    float w = (dx ? fx : 1.0f - fx) * (dy ? fy : 1.0f - fy);
                    sp += w * vb[(size_t)(yi * WW + xi) * CC];
                }
            }
        acc += aw[p] * sp;
    }
    sampled[(size_t)(b * HW + qi) * CC + h * HD + lane] = (_Float16)acc;
}

// ---------------------------------------------------------------------------
extern "C" void kernel_launch(void* const* d_in, const int* in_sizes, int n_in,
                              void* d_out, int out_size, void* d_ws, size_t ws_size,
                              hipStream_t stream) {
    const float* query  = (const float*)d_in[0];
    const float* value  = (const float*)d_in[1];
    const float* w_off  = (const float*)d_in[2];
    const float* b_off  = (const float*)d_in[3];
    const float* w_attn = (const float*)d_in[4];
    const float* b_attn = (const float*)d_in[5];
    const float* w_val  = (const float*)d_in[6];
    const float* b_val  = (const float*)d_in[7];
    const float* w_out  = (const float*)d_in[8];
    const float* b_out  = (const float*)d_in[9];
    float* out = (float*)d_out;

    int B = in_sizes[0] / (CC * HW);
    char* ws = (char*)d_ws;
    size_t o = 0;
    float*    qf32 = (float*)(ws + o);     o += (size_t)B * CC * HW * 4;
    _Float16* qf16 = (_Float16*)(ws + o);  o += (size_t)B * HW * CC * 2;
    _Float16* vf16 = (_Float16*)(ws + o);  o += (size_t)B * HW * CC * 2;
    _Float16* sampled = vf16;              // vf16 dead after value GEMM -> reuse
    float*    vbuf  = (float*)(ws + o);    o += (size_t)B * HW * CC * 4;
    float*    offb  = (float*)(ws + o);    o += (size_t)B * HW * 64 * 4;
    float*    attnb = (float*)(ws + o);    o += (size_t)B * HW * 32 * 4;
    _Float16* wvf   = (_Float16*)(ws + o); o += 256 * 256 * 2;
    _Float16* wof   = (_Float16*)(ws + o); o += 256 * 256 * 2;
    _Float16* woff  = (_Float16*)(ws + o); o += 256 * 64 * 2;
    _Float16* watt  = (_Float16*)(ws + o); o += 256 * 32 * 2;

    int total = B * CC * HW;
    prep_kernel<<<(total + 255) / 256, 256, 0, stream>>>(query, value, qf32, qf16, vf16, total);

    wconv_kernel<<<16, 256, 0, stream>>>(w_val,  wvf, 256);
    wconv_kernel<<<16, 256, 0, stream>>>(w_out,  wof, 256);
    wconv_kernel<<< 4, 256, 0, stream>>>(w_off,  woff, 64);
    wconv_kernel<<< 2, 256, 0, stream>>>(w_attn, watt, 32);

    int mtiles = (B * HW) / 16;   // 2048 for B=8
    dim3 blk(128);
    // value proj: N=256, 4 N-tiles per wave -> grid.y = 4
    gemm_kernel<4><<<dim3(mtiles / 4, 4), blk, 0, stream>>>(vf16, wvf, b_val, vbuf, 256);
    // offsets: N=64 -> one column group of 4 tiles
    gemm_kernel<4><<<dim3(mtiles / 4, 1), blk, 0, stream>>>(qf16, woff, b_off, offb, 64);
    // attn logits: N=32 -> one column group of 2 tiles
    gemm_kernel<2><<<dim3(mtiles / 4, 1), blk, 0, stream>>>(qf16, watt, b_attn, attnb, 32);

    sample_kernel<<<(B * HW * NHD) / 8, 256, 0, stream>>>(vbuf, offb, attnb, sampled);

    // output proj + fused (b_out + 2*q) epilogue straight into [B,C,H,W]
    gemm_final_kernel<4><<<dim3(mtiles / 4, 4), blk, 0, stream>>>(sampled, wof, b_out, qf32, out);
}
